// RWAGPU_31061203484816
// MI455X (gfx1250) — matrix operand order repaired and checked by reading, not compiled, not run on hardware
//
#include <hip/hip_runtime.h>
#include <hip/hip_bf16.h>

// ---------------------------------------------------------------------------
// Problem constants (match reference)
// ---------------------------------------------------------------------------
#define Bq   32
#define Tq   128
#define Fq   128
#define NFq  8
#define KWq  5
#define NCq  1024
#define NCLSq 1000
#define STATE_ELEMS (Bq * NFq * NCq)       // 262144
#define BN_COUNT (Bq * NCq)                // 32768 elements per channel per conv
#define KOUT (NFq * NCq)                   // 8192  (classifier K)

typedef __bf16 v16bf __attribute__((ext_vector_type(16)));
typedef float  v8f   __attribute__((ext_vector_type(8)));

__device__ __forceinline__ unsigned short f2bf_rne(float f) {
    unsigned u = __float_as_uint(f);
    unsigned r = u + 0x7fffu + ((u >> 16) & 1u);
    return (unsigned short)(r >> 16);
}
__device__ __forceinline__ unsigned pack2bf(float lo, float hi) {
    return ((unsigned)f2bf_rne(hi) << 16) | f2bf_rne(lo);
}

// ===========================================================================
// Fragment pre-pack kernels.  All WMMA operands are laid out in memory in the
// exact per-lane VGPR layout so GEMM inner loops are pure b128 loads + wmma.
//
// A-fragment (16x32 bf16), lane L: row = m0 + L%16, klo = (L/16)*16,
//   dwords 0..3 = K pairs (klo..klo+7), dwords 4..7 = K pairs (8+klo..+7).
// B-fragment (32x16 bf16), lane L: col = n0 + L%16, kb = (L/16)*16,
//   dword j = K pair (kb+2j, kb+2j+1).
// ===========================================================================

// ---- pack x [B,T,F] into A fragments, M index m = t*32+b, K = F = 128 -----
__global__ __launch_bounds__(256) void pack_x_kernel(const float* __restrict__ x,
                                                     unsigned* __restrict__ xap) {
    int tid = blockIdx.x * blockDim.x + threadIdx.x;   // 256*4*32 = 32768
    int row_tile = tid >> 7;
    int kc       = (tid >> 5) & 3;
    int lane     = tid & 31;
    int rowm = row_tile * 16 + (lane & 15);
    int t = rowm >> 5, b = rowm & 31;
    const float* arow = x + (size_t)(b * Tq + t) * Fq + kc * 32;
    int klo = (lane >> 4) * 16;
    unsigned* dst = xap + ((size_t)((row_tile * 4 + kc) * 32 + lane)) * 8;
#pragma unroll
    for (int j = 0; j < 4; ++j)
        dst[j] = pack2bf(arow[klo + 2 * j], arow[klo + 2 * j + 1]);
#pragma unroll
    for (int j = 0; j < 4; ++j)
        dst[4 + j] = pack2bf(arow[8 + klo + 2 * j], arow[8 + klo + 2 * j + 1]);
}

// ---- pack x_resize_w [NC,F] into B fragments: Bmat[k][n] = Wr[n][k] -------
__global__ __launch_bounds__(256) void pack_wr_kernel(const float* __restrict__ wr,
                                                      unsigned* __restrict__ wp) {
    int tid = blockIdx.x * blockDim.x + threadIdx.x;   // 64*4*32 = 8192
    int ntile = tid >> 7;
    int kc    = (tid >> 5) & 3;
    int lane  = tid & 31;
    int n  = ntile * 16 + (lane & 15);
    int kb = kc * 32 + ((lane >> 4) << 4);
    const float* row = wr + (size_t)n * Fq;
    unsigned* dst = wp + ((size_t)((ntile * 4 + kc) * 32 + lane)) * 8;
#pragma unroll
    for (int j = 0; j < 8; ++j)
        dst[j] = pack2bf(row[kb + 2 * j], row[kb + 2 * j + 1]);
}

// ---- pack o_w [NCLS, 8192] into B fragments (zero for n >= 1000) ----------
__global__ __launch_bounds__(256) void pack_ow_kernel(const float* __restrict__ ow,
                                                      unsigned* __restrict__ op) {
    int tid = blockIdx.x * blockDim.x + threadIdx.x;   // 63*256*32 = 516096
    int col_tile = tid >> 13;
    int kc       = (tid >> 5) & 255;
    int lane     = tid & 31;
    int n  = col_tile * 16 + (lane & 15);
    int kb = kc * 32 + ((lane >> 4) << 4);
    unsigned* dst = op + ((size_t)((col_tile * 256 + kc) * 32 + lane)) * 8;
    if (n < NCLSq) {
        const float* row = ow + (size_t)n * KOUT;
#pragma unroll
        for (int j = 0; j < 8; ++j)
            dst[j] = pack2bf(row[kb + 2 * j], row[kb + 2 * j + 1]);
    } else {
#pragma unroll
        for (int j = 0; j < 8; ++j) dst[j] = 0u;
    }
}

// ---- pack h_T [32, 8192] into A fragments ---------------------------------
__global__ __launch_bounds__(256) void pack_h_kernel(const float* __restrict__ h,
                                                     unsigned* __restrict__ hp) {
    int tid = blockIdx.x * blockDim.x + threadIdx.x;   // 2*256*32 = 16384
    int row_tile = tid >> 13;
    int kc       = (tid >> 5) & 255;
    int lane     = tid & 31;
    int rowm = row_tile * 16 + (lane & 15);
    const float* arow = h + (size_t)rowm * KOUT + kc * 32;
    int klo = (lane >> 4) * 16;
    unsigned* dst = hp + ((size_t)((row_tile * 256 + kc) * 32 + lane)) * 8;
#pragma unroll
    for (int j = 0; j < 4; ++j)
        dst[j] = pack2bf(arow[klo + 2 * j], arow[klo + 2 * j + 1]);
#pragma unroll
    for (int j = 0; j < 4; ++j)
        dst[4 + j] = pack2bf(arow[8 + klo + 2 * j], arow[8 + klo + 2 * j + 1]);
}

// ===========================================================================
// GEMM 1:  XR[t*B + b][c] = sum_k x[b][t][k] * Wr[c][k]
// M=4096, N=1024, K=128.  One wave per 16x16 tile, pure load+wmma loop.
// ===========================================================================
__global__ __launch_bounds__(128) void gemm_xr_kernel(const unsigned* __restrict__ xap,
                                                      const unsigned* __restrict__ wp,
                                                      float* __restrict__ xr) {
    int wave = threadIdx.x >> 5;
    int lane = threadIdx.x & 31;
    int tile = blockIdx.x * 4 + wave;        // 16384 tiles = 256 row * 64 col
    int row_tile = tile >> 6;
    int col_tile = tile & 63;

    const v16bf* ap = reinterpret_cast<const v16bf*>(
        xap + ((size_t)(row_tile * 4) * 32 + lane) * 8);
    const v16bf* bp = reinterpret_cast<const v16bf*>(
        wp + ((size_t)(col_tile * 4) * 32 + lane) * 8);

    v8f acc = {};
#pragma unroll
    for (int kc = 0; kc < 4; ++kc) {
        v16bf a = ap[kc * 32];               // stride 32 fragments = 32*32B
        v16bf b = bp[kc * 32];
        acc = __builtin_amdgcn_wmma_f32_16x16x32_bf16(
            false, a, false, b, (short)0, acc, false, false);
    }

    int n     = col_tile * 16 + (lane & 15);
    int mbase = row_tile * 16 + (lane >> 4) * 8;
#pragma unroll
    for (int j = 0; j < 8; ++j)
        xr[(size_t)(mbase + j) * NCq + n] = acc[j];
}

// ===========================================================================
// State init: place s / n / d / am into their d_out slots, h_slot = h + tanh(s)
// ===========================================================================
__global__ __launch_bounds__(256) void init_state_kernel(
    const float* __restrict__ s_in, const float* __restrict__ n_in,
    const float* __restrict__ d_in, const float* __restrict__ h_in,
    const float* __restrict__ am_in,
    float* __restrict__ s_o, float* __restrict__ n_o, float* __restrict__ d_o,
    float* __restrict__ h_o, float* __restrict__ am_o) {
    int i = blockIdx.x * blockDim.x + threadIdx.x;
    if (i >= STATE_ELEMS) return;
    float sv = s_in[i];
    s_o[i]  = sv;
    n_o[i]  = n_in[i];
    d_o[i]  = d_in[i];
    am_o[i] = am_in[i];
    h_o[i]  = h_in[i] + tanhf(sv);
}

// ===========================================================================
// Per-step conv kernel (g/u/a/dec pre-BN outputs + per-channel sum/sumsq)
// ===========================================================================
__global__ __launch_bounds__(256) void conv_step_kernel(
    int t,
    const float* __restrict__ xr_all,      // [T][B][NC]
    const float* __restrict__ hbuf,        // [B][NF][NC]
    const float* __restrict__ gw,  const float* __restrict__ gb,
    const float* __restrict__ uw,  const float* __restrict__ ub,
    const float* __restrict__ aw,  const float* __restrict__ ab,
    const float* __restrict__ dw,  const float* __restrict__ db,
    float* __restrict__ gbuf, float* __restrict__ ubuf,
    float* __restrict__ abuf, float* __restrict__ dbuf,
    float* __restrict__ stats) {           // [T][4][8][2]
    __shared__ float xh_s[9 * 260];
    __shared__ float wg_s[360], wa_s[360], wd_s[360], wu_s[40];
    __shared__ float bg_s[8], bu_s[8], ba_s[8], bd_s[8];
    __shared__ float stat_s[64];

    int tid = threadIdx.x;
    int b   = blockIdx.x >> 2;
    int c0  = (blockIdx.x & 3) * 256;

    for (int i = tid; i < 360; i += 256) { wg_s[i] = gw[i]; wa_s[i] = aw[i]; wd_s[i] = dw[i]; }
    for (int i = tid; i < 40;  i += 256) wu_s[i] = uw[i];
    if (tid < 8) { bg_s[tid] = gb[tid]; bu_s[tid] = ub[tid]; ba_s[tid] = ab[tid]; bd_s[tid] = db[tid]; }
    if (tid < 64) stat_s[tid] = 0.0f;

    const float* xr_t = xr_all + (size_t)(t * Bq + b) * NCq;
    for (int li = tid; li < 9 * 260; li += 256) {
        int ch = li / 260;
        int p  = li - ch * 260;
        int c  = c0 - 2 + p;
        float v = 0.0f;
        if (c >= 0 && c < NCq)
            v = (ch == 0) ? xr_t[c] : hbuf[(size_t)(b * NFq + (ch - 1)) * NCq + c];
        xh_s[li] = v;
    }
    __syncthreads();

    float xs[45];
#pragma unroll
    for (int i = 0; i < 9; ++i)
#pragma unroll
        for (int kw = 0; kw < KWq; ++kw)
            xs[i * 5 + kw] = xh_s[i * 260 + tid + kw];

    float acc[32];
#pragma unroll
    for (int o = 0; o < 8; ++o) {                 // g
        float a0 = bg_s[o];
        const float* w = &wg_s[o * 45];
#pragma unroll
        for (int q = 0; q < 45; ++q) a0 = fmaf(w[q], xs[q], a0);
        acc[o] = a0;
    }
#pragma unroll
    for (int o = 0; o < 8; ++o) {                 // u (xr channel only)
        float a0 = bu_s[o];
#pragma unroll
        for (int kw = 0; kw < 5; ++kw) a0 = fmaf(wu_s[o * 5 + kw], xs[kw], a0);
        acc[8 + o] = a0;
    }
#pragma unroll
    for (int o = 0; o < 8; ++o) {                 // a
        float a0 = ba_s[o];
        const float* w = &wa_s[o * 45];
#pragma unroll
        for (int q = 0; q < 45; ++q) a0 = fmaf(w[q], xs[q], a0);
        acc[16 + o] = a0;
    }
#pragma unroll
    for (int o = 0; o < 8; ++o) {                 // dec
        float a0 = bd_s[o];
        const float* w = &wd_s[o * 45];
#pragma unroll
        for (int q = 0; q < 45; ++q) a0 = fmaf(w[q], xs[q], a0);
        acc[24 + o] = a0;
    }

    int c = c0 + tid;
#pragma unroll
    for (int o = 0; o < 8; ++o) {
        size_t idx = (size_t)(b * NFq + o) * NCq + c;
        gbuf[idx] = acc[o];
        ubuf[idx] = acc[8 + o];
        abuf[idx] = acc[16 + o];
        dbuf[idx] = acc[24 + o];
    }

    int lane = tid & 31;
#pragma unroll
    for (int q = 0; q < 32; ++q) {
        float sv = acc[q];
        float qv = sv * sv;
#pragma unroll
        for (int off = 16; off > 0; off >>= 1) {
            sv += __shfl_down(sv, off, 32);
            qv += __shfl_down(qv, off, 32);
        }
        if (lane == 0) {
            atomicAdd(&stat_s[q * 2 + 0], sv);
            atomicAdd(&stat_s[q * 2 + 1], qv);
        }
    }
    __syncthreads();
    if (tid < 64) {
        int accIdx = tid >> 1;
        int sel    = tid & 1;
        int conv   = accIdx >> 3;
        int ch     = accIdx & 7;
        atomicAdd(&stats[t * 64 + conv * 16 + ch * 2 + sel], stat_s[tid]);
    }
}

// ===========================================================================
// Per-step BN + elementwise recurrent update (in place in d_out state slots)
// ===========================================================================
__global__ __launch_bounds__(256) void update_step_kernel(
    int t,
    const float* __restrict__ gbuf, const float* __restrict__ ubuf,
    const float* __restrict__ abuf, const float* __restrict__ dbuf,
    const float* __restrict__ stats,
    const float* __restrict__ g_gm, const float* __restrict__ g_bt,
    const float* __restrict__ u_gm, const float* __restrict__ u_bt,
    const float* __restrict__ a_gm, const float* __restrict__ a_bt,
    const float* __restrict__ d_gm, const float* __restrict__ d_bt,
    float* __restrict__ nbuf, float* __restrict__ dsbuf,
    float* __restrict__ hbuf, float* __restrict__ ambuf) {
    int idx = blockIdx.x * blockDim.x + threadIdx.x;
    if (idx >= STATE_ELEMS) return;
    int ch = (idx >> 10) & 7;

    const float inv = 1.0f / (float)BN_COUNT;
    const float* st = stats + t * 64;

    float gv = gbuf[idx], uv = ubuf[idx], av = abuf[idx], dv = dbuf[idx];

    float mu, var;
    mu  = st[0 * 16 + ch * 2] * inv;
    var = st[0 * 16 + ch * 2 + 1] * inv - mu * mu;
    float gbn = g_gm[ch] * (gv - mu) * rsqrtf(var + 1e-5f) + g_bt[ch];

    mu  = st[1 * 16 + ch * 2] * inv;
    var = st[1 * 16 + ch * 2 + 1] * inv - mu * mu;
    float ubn = u_gm[ch] * (uv - mu) * rsqrtf(var + 1e-5f) + u_bt[ch];

    mu  = st[2 * 16 + ch * 2] * inv;
    var = st[2 * 16 + ch * 2 + 1] * inv - mu * mu;
    float abn = a_gm[ch] * (av - mu) * rsqrtf(var + 1e-5f) + a_bt[ch];

    mu  = st[3 * 16 + ch * 2] * inv;
    var = st[3 * 16 + ch * 2 + 1] * inv - mu * mu;
    float dbn = d_gm[ch] * (dv - mu) * rsqrtf(var + 1e-5f) + d_bt[ch];

    float dec = 1.0f / (1.0f + __expf(-dbn));     // sigmoid
    float ed  = __expf(-dec);
    float z   = ubn * tanhf(gbn);

    float am  = ambuf[idx];
    float an  = fmaxf(am * ed, abn);
    float e1  = __expf(am - an);
    float e2  = __expf(abn - an);
    float nn  = nbuf[idx] * ed * e1 + z * e2;
    float dd  = dsbuf[idx] * ed * e1 + e2;

    nbuf[idx]  = nn;
    dsbuf[idx] = dd;
    ambuf[idx] = an;
    hbuf[idx]  = tanhf(nn / dd);
}

// ===========================================================================
// GEMM 2 (classifier): out[b][n] = h[b][:] . o_w[n][:] + o_b[n]
// M=32 (2 tiles) x N=1008 (63 tiles, guard n<1000) x K=8192 (256 chunks)
// Pure load+wmma inner loop over pre-packed fragments.
// ===========================================================================
__global__ __launch_bounds__(128) void gemm_out_kernel(
    const unsigned* __restrict__ hp, const unsigned* __restrict__ op,
    const float* __restrict__ ob, float* __restrict__ out) {
    int wave = threadIdx.x >> 5;
    int lane = threadIdx.x & 31;
    int tile = blockIdx.x * 4 + wave;
    if (tile >= 126) return;
    int row_tile = tile / 63;
    int col_tile = tile % 63;

    const v16bf* ap = reinterpret_cast<const v16bf*>(
        hp + ((size_t)(row_tile * 256) * 32 + lane) * 8);
    const v16bf* bp = reinterpret_cast<const v16bf*>(
        op + ((size_t)(col_tile * 256) * 32 + lane) * 8);

    v8f acc = {};
#pragma unroll 8
    for (int kc = 0; kc < 256; ++kc) {
        v16bf a = ap[kc * 32];
        v16bf b = bp[kc * 32];
        acc = __builtin_amdgcn_wmma_f32_16x16x32_bf16(
            false, a, false, b, (short)0, acc, false, false);
    }

    int n = col_tile * 16 + (lane & 15);
    if (n < NCLSq) {
        float bias = ob[n];
        int mbase = row_tile * 16 + (lane >> 4) * 8;
#pragma unroll
        for (int j = 0; j < 8; ++j)
            out[(size_t)(mbase + j) * NCLSq + n] = acc[j] + bias;
    }
}

// ===========================================================================
// Launch
// ===========================================================================
extern "C" void kernel_launch(void* const* d_in, const int* in_sizes, int n_in,
                              void* d_out, int out_size, void* d_ws, size_t ws_size,
                              hipStream_t stream) {
    (void)in_sizes; (void)n_in; (void)out_size; (void)ws_size;

    const float* x      = (const float*)d_in[0];
    const float* s_in   = (const float*)d_in[1];
    const float* n_in_p = (const float*)d_in[2];
    const float* d_in_p = (const float*)d_in[3];
    const float* h_in_p = (const float*)d_in[4];
    const float* am_in  = (const float*)d_in[5];
    const float* wr     = (const float*)d_in[6];
    const float* gw = (const float*)d_in[7],  *gb = (const float*)d_in[8];
    const float* ggm = (const float*)d_in[9], *gbt = (const float*)d_in[10];
    const float* uw = (const float*)d_in[11], *ub = (const float*)d_in[12];
    const float* ugm = (const float*)d_in[13], *ubt = (const float*)d_in[14];
    const float* aw = (const float*)d_in[15], *ab = (const float*)d_in[16];
    const float* agm = (const float*)d_in[17], *abt = (const float*)d_in[18];
    const float* dw = (const float*)d_in[19], *db = (const float*)d_in[20];
    const float* dgm = (const float*)d_in[21], *dbt = (const float*)d_in[22];
    const float* owp = (const float*)d_in[23];
    const float* obp = (const float*)d_in[24];

    float* out = (float*)d_out;
    // output layout: outs | s | n | d | h | am
    float* s_o  = out + 32000;
    float* n_o  = s_o  + STATE_ELEMS;
    float* d_o  = n_o  + STATE_ELEMS;
    float* h_o  = d_o  + STATE_ELEMS;
    float* am_o = h_o  + STATE_ELEMS;

    // workspace layout (in 4-byte units)
    unsigned* wsu = (unsigned*)d_ws;
    float*    XR     = (float*)wsu;                       //  4,194,304 f
    unsigned* wpack  = wsu + 4194304;                     //     65,536 dw
    unsigned* xapack = wpack + 65536;                     //    262,144 dw
    unsigned* owpack = xapack + 262144;                   //  4,128,768 dw
    unsigned* hapack = owpack + 4128768;                  //    131,072 dw
    float*    gbuf   = (float*)(hapack + 131072);         //    262,144 f
    float*    ubuf   = gbuf + STATE_ELEMS;
    float*    abuf   = ubuf + STATE_ELEMS;
    float*    dbuf   = abuf + STATE_ELEMS;
    float*    stats  = dbuf + STATE_ELEMS;                //      8,192 f

    hipMemsetAsync(stats, 0, (size_t)Tq * 64 * sizeof(float), stream);

    pack_wr_kernel<<<dim3(32),   dim3(256), 0, stream>>>(wr, wpack);
    pack_x_kernel <<<dim3(128),  dim3(256), 0, stream>>>(x, xapack);
    pack_ow_kernel<<<dim3(2016), dim3(256), 0, stream>>>(owp, owpack);

    gemm_xr_kernel<<<dim3(4096), dim3(128), 0, stream>>>(xapack, wpack, XR);
    init_state_kernel<<<dim3(STATE_ELEMS / 256), dim3(256), 0, stream>>>(
        s_in, n_in_p, d_in_p, h_in_p, am_in, s_o, n_o, d_o, h_o, am_o);

    for (int t = 0; t < Tq; ++t) {
        conv_step_kernel<<<dim3(128), dim3(256), 0, stream>>>(
            t, XR, h_o, gw, gb, uw, ub, aw, ab, dw, db,
            gbuf, ubuf, abuf, dbuf, stats);
        update_step_kernel<<<dim3(STATE_ELEMS / 256), dim3(256), 0, stream>>>(
            t, gbuf, ubuf, abuf, dbuf, stats,
            ggm, gbt, ugm, ubt, agm, abt, dgm, dbt,
            n_o, d_o, h_o, am_o);
    }

    pack_h_kernel<<<dim3(64), dim3(256), 0, stream>>>(h_o, hapack);
    gemm_out_kernel<<<dim3(32), dim3(128), 0, stream>>>(hapack, owpack, obp, out);
}
